// GridGatingUnit_10720238371185
// MI455X (gfx1250) — matrix operand order repaired and checked by reading, not compiled, not run on hardware
//
#include <hip/hip_runtime.h>
#include <hip/hip_bf16.h>

// ---------------------------------------------------------------------------
// GridGatingUnit for MI455X (gfx1250, wave32)
//   out[b,k,m,c] = u[b,k,m,c] * ( sum_n LN(v)[b,n,m,c] * W[k,n] + bias[k] + 1 )
// Strategy:
//   - LayerNorm kernel writes normalized V as bf16 directly in WMMA B-fragment
//     register-image layout into d_ws (128 MB -> resident in 192 MB L2).
//   - Tiny kernel pre-swizzles W into WMMA A-fragment layout (bf16).
//   - GEMM kernel: v_wmma_f32_16x16x32_bf16, K=256 in 8 steps, fp32 acc,
//     fused gating epilogue.
// ---------------------------------------------------------------------------

typedef __bf16 v16bf __attribute__((ext_vector_type(16)));
typedef float  v8f   __attribute__((ext_vector_type(8)));

#define B_    4
#define N_    256   // k and n extent (W is N_ x N_)
#define M_    256
#define C_    256
#define JTOT  (M_ * C_)          // 65536 columns of the GEMM per batch
#define KBLKS (N_ / 16)          // 16 k-row blocks
#define NBLKS (N_ / 32)          // 8 K(reduction)-fragments
#define JBLKS (JTOT / 16)        // 4096 j-tiles
#define EPS_  1e-5f

// fp32 -> bf16, round-to-nearest-even (matches v_cvt semantics closely enough)
static __device__ __forceinline__ unsigned short f2bf(float f) {
    unsigned int u = __builtin_bit_cast(unsigned int, f);
    u += 0x7FFFu + ((u >> 16) & 1u);
    return (unsigned short)(u >> 16);
}

// ---------------------------------------------------------------------------
// Kernel 0: swizzle W [k][n] fp32 -> bf16 A-fragments.
// A-frag (16x32 bf16, ISA 7.12.2): M = lane&15,
//   K = (v>=4?16:0) + (lane>>4)*8 + (v&3)*2 + h   (v = VGPR 0..7, h = half)
// Storage: frag f = kblk*NBLKS + nblk; lane-major, element e = lane*16 + v*2 + h.
// ---------------------------------------------------------------------------
__global__ __launch_bounds__(256) void ggu_wconv(const float* __restrict__ W,
                                                 unsigned int* __restrict__ Wf32) {
    const int f    = blockIdx.x;          // 0..127
    const int kblk = f >> 3;
    const int nblk = f & 7;
    const int tid  = threadIdx.x;         // 0..255 -> dword e2 = tid
    const int lane = tid >> 3;            // e = tid*2 -> lane = e>>4
    const int v    = tid & 7;             // ((tid*2)&15)>>1
    const int k    = kblk * 16 + (lane & 15);
    const int nbase = nblk * 32 + ((v >> 2) << 4) + ((lane >> 4) << 3) + ((v & 3) << 1);
    const unsigned short lo = f2bf(W[k * N_ + nbase + 0]);
    const unsigned short hi = f2bf(W[k * N_ + nbase + 1]);
    Wf32[f * 256 + tid] = (unsigned int)lo | ((unsigned int)hi << 16);
}

// ---------------------------------------------------------------------------
// Kernel 1: LayerNorm over C for two adjacent n-rows, emit bf16 into
// B-fragment layout.
// B-frag (32x16 bf16): N = lane&15, K = (v>=4?16:0) + (lane>>4)*8 + (v&3)*2 + h.
// frag = (b*NBLKS + kb)*JBLKS + jb ; element e = lane*16 + v*2 + h.
// Rows n0 (even) / n0+1 share (lane,v) and differ in h -> packed dword store.
// ---------------------------------------------------------------------------
__global__ __launch_bounds__(256) void ggu_lnorm(const float* __restrict__ X,
                                                 const float* __restrict__ nw,
                                                 const float* __restrict__ nb,
                                                 unsigned int* __restrict__ Vf32) {
    const int bx = blockIdx.x;
    const int m     = bx & (M_ - 1);
    const int npair = (bx >> 8) & 127;
    const int b     = bx >> 15;
    const int n0    = npair * 2;
    const int c     = threadIdx.x;

    const size_t row0 = ((size_t)((b * N_ + n0) * M_ + m)) * (2 * C_) + C_;
    const size_t row1 = row0 + (size_t)M_ * (2 * C_);
    const float v0 = X[row0 + c];
    const float v1 = X[row1 + c];

    // block reduction: sum and sum-of-squares for both rows
    float s0 = v0, s1 = v1, q0 = v0 * v0, q1 = v1 * v1;
    #pragma unroll
    for (int off = 16; off > 0; off >>= 1) {
        s0 += __shfl_xor(s0, off, 32);
        s1 += __shfl_xor(s1, off, 32);
        q0 += __shfl_xor(q0, off, 32);
        q1 += __shfl_xor(q1, off, 32);
    }
    __shared__ float red[4][8];
    const int wave = threadIdx.x >> 5;
    if ((threadIdx.x & 31) == 0) {
        red[0][wave] = s0; red[1][wave] = s1; red[2][wave] = q0; red[3][wave] = q1;
    }
    __syncthreads();
    float S0 = 0.f, S1 = 0.f, Q0 = 0.f, Q1 = 0.f;
    #pragma unroll
    for (int i = 0; i < 8; ++i) {
        S0 += red[0][i]; S1 += red[1][i]; Q0 += red[2][i]; Q1 += red[3][i];
    }
    const float inv_c = 1.0f / (float)C_;
    const float mu0 = S0 * inv_c, mu1 = S1 * inv_c;
    const float r0 = rsqrtf(Q0 * inv_c - mu0 * mu0 + EPS_);
    const float r1 = rsqrtf(Q1 * inv_c - mu1 * mu1 + EPS_);
    const float w = nw[c], bb = nb[c];
    const float y0 = (v0 - mu0) * r0 * w + bb;
    const float y1 = (v1 - mu1) * r1 * w + bb;

    // scatter into B-fragment layout (lands in L2; V workspace is L2-resident)
    const int lane = ((n0 >> 3) & 1) * 16 + (c & 15);
    const int vreg = ((n0 >> 4) & 1) * 4 + ((n0 & 7) >> 1);
    const int jb   = m * 16 + (c >> 4);
    const int kb   = n0 >> 5;
    const int frag = (b * NBLKS + kb) * JBLKS + jb;
    Vf32[(size_t)frag * 256 + lane * 8 + vreg] =
        (unsigned int)f2bf(y0) | ((unsigned int)f2bf(y1) << 16);
}

// ---------------------------------------------------------------------------
// Kernel 2: GEMM + gating epilogue. One wave = one 16(k)x16(j) tile,
// 4 j-tiles per wave with all 8 A-fragments held in registers.
// grid: b(4) * kblk(16) * jgrp(128) ; block = 256 threads = 8 waves.
// ---------------------------------------------------------------------------
__global__ __launch_bounds__(256) void ggu_gemm(const unsigned short* __restrict__ Wf,
                                                const unsigned short* __restrict__ Vf,
                                                const float* __restrict__ X,
                                                const float* __restrict__ bias,
                                                float* __restrict__ Out) {
    const int bx   = blockIdx.x;
    const int jgrp = bx & 127;
    const int kblk = (bx >> 7) & 15;
    const int b    = bx >> 11;
    const int wave = threadIdx.x >> 5;
    const int lane = threadIdx.x & 31;

    // A fragments for this k-row block: 8 x v16bf (64 VGPRs), reused 4x
    v16bf afrag[NBLKS];
    #pragma unroll
    for (int kk = 0; kk < NBLKS; ++kk) {
        const v16bf* pa =
            (const v16bf*)(Wf + (size_t)(kblk * NBLKS + kk) * 512);
        afrag[kk] = pa[lane];
    }

    // per-lane bias row (k = k0 + (lane>=16)*8 + a)
    const int k0   = kblk * 16;
    const int krow = k0 + ((lane >> 4) << 3);
    float biasv[8];
    #pragma unroll
    for (int a = 0; a < 8; ++a) biasv[a] = bias[krow + a] + 1.0f;

    const int jb0 = (jgrp * 8 + wave) * 4;
    #pragma unroll
    for (int t = 0; t < 4; ++t) {
        const int jb = jb0 + t;
        v8f acc = {0.f, 0.f, 0.f, 0.f, 0.f, 0.f, 0.f, 0.f};
        #pragma unroll
        for (int kk = 0; kk < NBLKS; ++kk) {
            const v16bf* pb =
                (const v16bf*)(Vf + ((size_t)((b * NBLKS + kk) * JBLKS + jb)) * 512);
            const v16bf bfrag = pb[lane];
            acc = __builtin_amdgcn_wmma_f32_16x16x32_bf16(
                /*neg_a=*/false, afrag[kk],
                /*neg_b=*/false, bfrag,
                /*c_mod=*/(short)0, acc,
                /*reuse_a=*/false, /*reuse_b=*/false);
        }
        // epilogue: out[b,k,m,c] = u * (acc + bias[k] + 1)
        const int j = jb * 16 + (lane & 15);
        const int m = j >> 8;
        const int c = j & (C_ - 1);
        #pragma unroll
        for (int a = 0; a < 8; ++a) {
            const int k = krow + a;
            const size_t ubase = ((size_t)((b * N_ + k) * M_ + m)) * (2 * C_) + c;
            const float uval = X[ubase];
            const size_t obase = ((size_t)(b * N_ + k)) * JTOT + j;
            Out[obase] = uval * (acc[a] + biasv[a]);
        }
    }
}

extern "C" void kernel_launch(void* const* d_in, const int* in_sizes, int n_in,
                              void* d_out, int out_size, void* d_ws, size_t ws_size,
                              hipStream_t stream) {
    const float* x      = (const float*)d_in[0];
    const float* norm_w = (const float*)d_in[1];
    const float* norm_b = (const float*)d_in[2];
    const float* W      = (const float*)d_in[3];
    const float* bias   = (const float*)d_in[4];
    float* out = (float*)d_out;

    // workspace: [0, 128 KiB) = W bf16 A-fragments ; then 128 MiB V bf16 B-fragments
    unsigned char* ws = (unsigned char*)d_ws;
    unsigned int*   Wf32 = (unsigned int*)ws;
    unsigned short* Wf   = (unsigned short*)ws;
    unsigned int*   Vf32 = (unsigned int*)(ws + 131072);
    unsigned short* Vf   = (unsigned short*)(ws + 131072);

    // 1) W -> bf16 A-fragment layout (128 fragments)
    ggu_wconv<<<KBLKS * NBLKS, 256, 0, stream>>>(W, Wf32);

    // 2) LayerNorm + bf16 B-fragment scatter (b * n-pairs * m blocks)
    ggu_lnorm<<<B_ * (N_ / 2) * M_, 256, 0, stream>>>(x, norm_w, norm_b, Vf32);

    // 3) bf16 WMMA GEMM + gating epilogue
    ggu_gemm<<<B_ * KBLKS * 128, 256, 0, stream>>>(Wf, Vf, x, bias, out);
}
